// AttentionBlock_1331439861975
// MI455X (gfx1250) — matrix operand order repaired and checked by reading, not compiled, not run on hardware
//
#include <hip/hip_runtime.h>
#include <hip/hip_bf16.h>

// ---------------------------------------------------------------------------
// Problem constants (from reference)
// ---------------------------------------------------------------------------
#define B_  8
#define C_  256
#define H_  48
#define W_  48
#define N_  (H_ * W_)        // 2304 pixels per batch
#define G_  32               // groups
#define CPG (C_ / G_)        // 8 channels per group
#define CC  (C_ * C_)        // 65536
#define EPS 1e-5f

// ---------------------------------------------------------------------------
// WMMA vector types + helpers (CDNA5 wave32, v_wmma_f32_16x16x32_f16)
// ---------------------------------------------------------------------------
typedef __attribute__((ext_vector_type(16))) _Float16 v16h;
typedef __attribute__((ext_vector_type(8)))  _Float16 v8h;
typedef __attribute__((ext_vector_type(8)))  float    v8f;

// A-matrix (16x32 f16) fragment load. `base` points at element (row0, k0) of a
// row-major matrix with leading dimension `lda` (elements). Per ISA layout:
// lane m=lane&15 holds row m; lane group g=lane>>4 holds K = {8g..8g+7} in
// slots 0..7 and K = {16+8g..16+8g+7} in slots 8..15.
__device__ __forceinline__ v16h load_a_frag(const _Float16* base, int lda, int lane) {
    int m = lane & 15, g = lane >> 4;
    const _Float16* p = base + (size_t)m * lda + g * 8;
    v8h lo = *(const v8h*)(p);       // K = k0 + 8g + [0..7]
    v8h hi = *(const v8h*)(p + 16);  // K = k0 + 16 + 8g + [0..7]
    v16h r;
#pragma unroll
    for (int i = 0; i < 8; ++i) { r[i] = lo[i]; r[i + 8] = hi[i]; }
    return r;
}

// B-matrix (32x16 f16) fragment load. Columns of B are stored contiguously:
// `base` points at (k0, col0) where column c lives at base + c*ldb, K
// contiguous. Lane n=lane&15 holds column n; group g holds K = {8g..8g+7} in
// slots 0..7 and K = {16+8g..16+8g+7} in slots 8..15.
__device__ __forceinline__ v16h load_b_frag(const _Float16* base, int ldb, int lane) {
    int n = lane & 15, g = lane >> 4;
    const _Float16* p = base + (size_t)n * ldb + g * 8;
    v8h lo = *(const v8h*)(p);
    v8h hi = *(const v8h*)(p + 16);
    v16h r;
#pragma unroll
    for (int i = 0; i < 8; ++i) { r[i] = lo[i]; r[i + 8] = hi[i]; }
    return r;
}

__device__ __forceinline__ v8f wmma_f32_f16(v16h a, v16h b, v8f c) {
    return __builtin_amdgcn_wmma_f32_16x16x32_f16(
        /*neg_a=*/false, a, /*neg_b=*/false, b,
        /*c_mod=*/(short)0, c, /*reuse_a=*/false, /*reuse_b=*/false);
}

// CDNA5 async copy: global -> LDS, 16 bytes, tracked by ASYNCcnt.
// vdst = VGPR with LDS byte offset, vaddr = 64-bit global address.
__device__ __forceinline__ void async_copy_b128(unsigned lds_off, const void* gptr) {
    asm volatile("global_load_async_to_lds_b128 %0, %1, off"
                 :: "v"(lds_off), "v"(gptr)
                 : "memory");
}
__device__ __forceinline__ void wait_asynccnt0() {
    asm volatile("s_wait_asynccnt 0x0" ::: "memory");
}

// ---------------------------------------------------------------------------
// Kernel 1: GroupNorm statistics. One block per (b, g): reduce 8*2304
// contiguous floats -> mean, rstd.
// ---------------------------------------------------------------------------
__global__ void gn_stats_kernel(const float* __restrict__ x,
                                float* __restrict__ mean,
                                float* __restrict__ rstd) {
    const int bg = blockIdx.x;                 // b*G + g
    const float* p = x + (size_t)bg * (CPG * N_);
    float s = 0.f, sq = 0.f;
    for (int i = threadIdx.x; i < CPG * N_; i += blockDim.x) {
        float v = p[i];
        s += v; sq += v * v;
    }
    __shared__ float ss[256], ssq[256];
    int t = threadIdx.x;
    ss[t] = s; ssq[t] = sq;
    __syncthreads();
    for (int st = 128; st > 0; st >>= 1) {
        if (t < st) { ss[t] += ss[t + st]; ssq[t] += ssq[t + st]; }
        __syncthreads();
    }
    if (t == 0) {
        const float cnt = (float)(CPG * N_);
        float m   = ss[0] / cnt;
        float var = ssq[0] / cnt - m * m;
        mean[bg] = m;
        rstd[bg] = rsqrtf(var + EPS);
    }
}

// ---------------------------------------------------------------------------
// Kernel 2: apply GroupNorm affine and transpose [B][C][N] f32 ->
// [B][N][C] f16 via an LDS tile (64 pixels x 32 channels per block).
// ---------------------------------------------------------------------------
__global__ void gn_apply_kernel(const float* __restrict__ x,
                                const float* __restrict__ gnw,
                                const float* __restrict__ gnb,
                                const float* __restrict__ mean,
                                const float* __restrict__ rstd,
                                _Float16* __restrict__ xn16) {
    __shared__ _Float16 tile[32][66];          // +2 pad -> 33-bank stride
    const int b  = blockIdx.z;
    const int c0 = blockIdx.y * 32;
    const int n0 = blockIdx.x * 64;
    const int t  = threadIdx.x;

    // Read phase: coalesced over pixels.
    const int pl = t & 63;
    const int cq = t >> 6;                     // 4 channel rows per pass
#pragma unroll
    for (int cc = 0; cc < 8; ++cc) {
        int cl = cc * 4 + cq;
        int c  = c0 + cl;
        int bg = b * G_ + (c >> 3);            // CPG == 8
        float v = x[((size_t)(b * C_ + c)) * N_ + n0 + pl];
        v = (v - mean[bg]) * rstd[bg] * gnw[c] + gnb[c];
        tile[cl][pl] = (_Float16)v;
    }
    __syncthreads();

    // Write phase: coalesced over channels (pixel-major f16 layout).
    const int cl2 = t & 31;
    const int pq  = t >> 5;                    // 8 pixels per pass
#pragma unroll
    for (int pp = 0; pp < 8; ++pp) {
        int pl2 = pp * 8 + pq;
        xn16[((size_t)b * N_ + n0 + pl2) * C_ + c0 + cl2] = tile[cl2][pl2];
    }
}

// ---------------------------------------------------------------------------
// Kernel 3: convert Wq/Wk/Wv (row-major [C_out][C_in] f32) to f16.
// ---------------------------------------------------------------------------
__global__ void cvt_w_kernel(const float* __restrict__ wq,
                             const float* __restrict__ wk,
                             const float* __restrict__ wv,
                             _Float16* __restrict__ w16) {
    int i = blockIdx.x * blockDim.x + threadIdx.x;   // 0..CC-1
    w16[i]          = (_Float16)wq[i];
    w16[CC + i]     = (_Float16)wk[i];
    w16[2 * CC + i] = (_Float16)wv[i];
}

// ---------------------------------------------------------------------------
// Kernel 4: QKV projection GEMMs with 4x4 register blocking (64x64 per wave,
// 4 waves arranged 2x2 -> 128(pix) x 128(cout) per block).
// q/k stored pixel-major [B][N][C] f16, v stored channel-major [B][C][N] f16.
// Grid (N/128, C/128, B*3), block 128.
// ---------------------------------------------------------------------------
__global__ void __launch_bounds__(128)
qkv_gemm_kernel(const _Float16* __restrict__ xn16,
                const _Float16* __restrict__ w16,
                const float* __restrict__ bq,
                const float* __restrict__ bk,
                const float* __restrict__ bv,
                _Float16* __restrict__ q16,
                _Float16* __restrict__ k16,
                _Float16* __restrict__ v16) {
    const int lane = threadIdx.x & 31;
    const int wave = threadIdx.x >> 5;
    const int mat  = blockIdx.z % 3;           // 0=q 1=k 2=v
    const int b    = blockIdx.z / 3;
    const int pix0 = blockIdx.x * 128 + (wave >> 1) * 64;
    const int o0   = blockIdx.y * 128 + (wave & 1) * 64;

    const _Float16* W     = w16 + (size_t)mat * CC;
    const float*    bias  = (mat == 0) ? bq : (mat == 1 ? bk : bv);
    const _Float16* Abase = xn16 + ((size_t)b * N_ + pix0) * C_;
    const _Float16* Bbase = W + (size_t)o0 * C_;   // column o == row o of W

    const v8f zacc = {};
    v8f acc[4][4];
#pragma unroll
    for (int i = 0; i < 4; ++i)
#pragma unroll
        for (int j = 0; j < 4; ++j) acc[i][j] = zacc;

    for (int kc = 0; kc < C_; kc += 32) {
        v16h a[4], bb[4];
#pragma unroll
        for (int i = 0; i < 4; ++i)
            a[i] = load_a_frag(Abase + (size_t)(16 * i) * C_ + kc, C_, lane);
#pragma unroll
        for (int j = 0; j < 4; ++j)
            bb[j] = load_b_frag(Bbase + (size_t)(16 * j) * C_ + kc, C_, lane);
#pragma unroll
        for (int i = 0; i < 4; ++i)
#pragma unroll
            for (int j = 0; j < 4; ++j)
                acc[i][j] = wmma_f32_f16(a[i], bb[j], acc[i][j]);
    }

#pragma unroll
    for (int i = 0; i < 4; ++i) {
        const int rbase = pix0 + 16 * i + 8 * (lane >> 4);
#pragma unroll
        for (int j = 0; j < 4; ++j) {
            const int   col  = o0 + 16 * j + (lane & 15);
            const float bval = bias[col];
            if (mat < 2) {
                _Float16* dst = (mat == 0) ? q16 : k16;
#pragma unroll
                for (int r = 0; r < 8; ++r)
                    dst[((size_t)b * N_ + rbase + r) * C_ + col] =
                        (_Float16)(acc[i][j][r] + bval);
            } else {
#pragma unroll
                for (int r = 0; r < 8; ++r)
                    v16[((size_t)b * C_ + col) * N_ + rbase + r] =
                        (_Float16)(acc[i][j][r] + bval);
            }
        }
    }
}

// ---------------------------------------------------------------------------
// Kernel 5: energy GEMM  E[b][n][m] = (sum_c Q[n][c] K[m][c]) / 16.
// 4x4 register blocking, 64x64 per wave, 128x128 per block.
// Grid (N/128, N/128, B), block 128.
// ---------------------------------------------------------------------------
__global__ void __launch_bounds__(128)
energy_gemm_kernel(const _Float16* __restrict__ q16,
                   const _Float16* __restrict__ k16,
                   float* __restrict__ E) {
    const int lane = threadIdx.x & 31;
    const int wave = threadIdx.x >> 5;
    const int b    = blockIdx.z;
    const int n0   = blockIdx.x * 128 + (wave >> 1) * 64;
    const int m0   = blockIdx.y * 128 + (wave & 1) * 64;

    const _Float16* Abase = q16 + ((size_t)b * N_ + n0) * C_;
    const _Float16* Bbase = k16 + ((size_t)b * N_ + m0) * C_;

    const v8f zacc = {};
    v8f acc[4][4];
#pragma unroll
    for (int i = 0; i < 4; ++i)
#pragma unroll
        for (int j = 0; j < 4; ++j) acc[i][j] = zacc;

    for (int kc = 0; kc < C_; kc += 32) {
        v16h a[4], bb[4];
#pragma unroll
        for (int i = 0; i < 4; ++i)
            a[i] = load_a_frag(Abase + (size_t)(16 * i) * C_ + kc, C_, lane);
#pragma unroll
        for (int j = 0; j < 4; ++j)
            bb[j] = load_b_frag(Bbase + (size_t)(16 * j) * C_ + kc, C_, lane);
#pragma unroll
        for (int i = 0; i < 4; ++i)
#pragma unroll
            for (int j = 0; j < 4; ++j)
                acc[i][j] = wmma_f32_f16(a[i], bb[j], acc[i][j]);
    }

    const float scale = 0.0625f;               // 1/sqrt(256)
#pragma unroll
    for (int i = 0; i < 4; ++i) {
        const int rb = n0 + 16 * i + 8 * (lane >> 4);
#pragma unroll
        for (int j = 0; j < 4; ++j) {
            const int m = m0 + 16 * j + (lane & 15);
#pragma unroll
            for (int r = 0; r < 8; ++r)
                E[((size_t)b * N_ + rb + r) * N_ + m] = acc[i][j][r] * scale;
        }
    }
}

// ---------------------------------------------------------------------------
// Kernel 6: row softmax.  Row is DMA'd into LDS with CDNA5 async loads
// (global_load_async_to_lds_b128, ASYNCcnt), reductions run from LDS.
// One block (256 thr) per row of 2304 = 9 elems/thread.
// ---------------------------------------------------------------------------
__global__ void softmax_kernel(const float* __restrict__ E,
                               _Float16* __restrict__ P) {
    __shared__ float row_sh[N_];               // 9216 B
    __shared__ float red[256];
    const size_t row = blockIdx.x;             // b*N + n
    const float* src = E + row * N_;
    _Float16*    dst = P + row * N_;
    const int t = threadIdx.x;

    // Async DMA the row into LDS: 576 x 16-byte chunks across 256 threads.
    // Generic shared pointers carry the LDS byte offset in their low 32 bits.
    const unsigned lds_base = (unsigned)(size_t)(&row_sh[0]);
    for (int chunk = t; chunk < N_ / 4; chunk += 256) {
        async_copy_b128(lds_base + (unsigned)chunk * 16u, src + chunk * 4);
    }
    wait_asynccnt0();                          // s_wait_asynccnt 0
    __syncthreads();

    float vals[9];
    float mx = -3.0e38f;
#pragma unroll
    for (int i = 0; i < 9; ++i) {
        vals[i] = row_sh[t + i * 256];
        mx = fmaxf(mx, vals[i]);
    }
    red[t] = mx;
    __syncthreads();
    for (int s = 128; s > 0; s >>= 1) {
        if (t < s) red[t] = fmaxf(red[t], red[t + s]);
        __syncthreads();
    }
    mx = red[0];
    __syncthreads();

    float sum = 0.f;
#pragma unroll
    for (int i = 0; i < 9; ++i) {
        vals[i] = __expf(vals[i] - mx);
        sum += vals[i];
    }
    red[t] = sum;
    __syncthreads();
    for (int s = 128; s > 0; s >>= 1) {
        if (t < s) red[t] += red[t + s];
        __syncthreads();
    }
    const float inv = 1.0f / red[0];
#pragma unroll
    for (int i = 0; i < 9; ++i)
        dst[t + i * 256] = (_Float16)(vals[i] * inv);
}

// ---------------------------------------------------------------------------
// Kernel 7: output GEMM  out[b][c][n] = sum_m P[n][m] * V[c][m]  (fp32 out).
// 4x4 register blocking, 64x64 per wave, 128(n) x 128(c) per block.
// Grid (N/128, C/128, B), block 128.  72 K-steps over m.
// ---------------------------------------------------------------------------
__global__ void __launch_bounds__(128)
out_gemm_kernel(const _Float16* __restrict__ P,
                const _Float16* __restrict__ v16,
                float* __restrict__ out) {
    const int lane = threadIdx.x & 31;
    const int wave = threadIdx.x >> 5;
    const int b    = blockIdx.z;
    const int n0   = blockIdx.x * 128 + (wave >> 1) * 64;
    const int c0   = blockIdx.y * 128 + (wave & 1) * 64;

    const _Float16* Abase = P   + ((size_t)b * N_ + n0) * N_;
    const _Float16* Bbase = v16 + ((size_t)b * C_ + c0) * N_;

    const v8f zacc = {};
    v8f acc[4][4];
#pragma unroll
    for (int i = 0; i < 4; ++i)
#pragma unroll
        for (int j = 0; j < 4; ++j) acc[i][j] = zacc;

    for (int m = 0; m < N_; m += 32) {
        v16h a[4], bb[4];
#pragma unroll
        for (int i = 0; i < 4; ++i)
            a[i] = load_a_frag(Abase + (size_t)(16 * i) * N_ + m, N_, lane);
#pragma unroll
        for (int j = 0; j < 4; ++j)
            bb[j] = load_b_frag(Bbase + (size_t)(16 * j) * N_ + m, N_, lane);
#pragma unroll
        for (int i = 0; i < 4; ++i)
#pragma unroll
            for (int j = 0; j < 4; ++j)
                acc[i][j] = wmma_f32_f16(a[i], bb[j], acc[i][j]);
    }

#pragma unroll
    for (int i = 0; i < 4; ++i) {
        const int rb = n0 + 16 * i + 8 * (lane >> 4);
#pragma unroll
        for (int j = 0; j < 4; ++j) {
            const int c = c0 + 16 * j + (lane & 15);
#pragma unroll
            for (int r = 0; r < 8; ++r)
                out[((size_t)b * C_ + c) * N_ + rb + r] = acc[i][j][r];
        }
    }
}

// ---------------------------------------------------------------------------
// Host launcher
// ---------------------------------------------------------------------------
extern "C" void kernel_launch(void* const* d_in, const int* in_sizes, int n_in,
                              void* d_out, int out_size, void* d_ws, size_t ws_size,
                              hipStream_t stream) {
    const float* input = (const float*)d_in[0];
    const float* Wq    = (const float*)d_in[1];
    const float* bq    = (const float*)d_in[2];
    const float* Wk    = (const float*)d_in[3];
    const float* bk    = (const float*)d_in[4];
    const float* Wv    = (const float*)d_in[5];
    const float* bv    = (const float*)d_in[6];
    const float* gn_w  = (const float*)d_in[7];
    const float* gn_b  = (const float*)d_in[8];
    float*       out   = (float*)d_out;

    // Workspace carve-up (256-B aligned sections).
    char*  ws  = (char*)d_ws;
    size_t off = 0;
    auto carve = [&](size_t bytes) -> char* {
        char* p = ws + off;
        off = (off + bytes + 255) & ~(size_t)255;
        return p;
    };
    float*    mean = (float*)carve(B_ * G_ * sizeof(float));          //   1 KB
    float*    rstd = (float*)carve(B_ * G_ * sizeof(float));          //   1 KB
    _Float16* w16  = (_Float16*)carve((size_t)3 * CC * 2);            // 384 KB
    _Float16* xn16 = (_Float16*)carve((size_t)B_ * N_ * C_ * 2);      // 9.4 MB
    _Float16* q16  = (_Float16*)carve((size_t)B_ * N_ * C_ * 2);      // 9.4 MB
    _Float16* k16  = (_Float16*)carve((size_t)B_ * N_ * C_ * 2);      // 9.4 MB
    _Float16* v16  = (_Float16*)carve((size_t)B_ * C_ * N_ * 2);      // 9.4 MB
    _Float16* P16  = (_Float16*)carve((size_t)B_ * N_ * N_ * 2);      //  85 MB
    float*    E32  = (float*)carve((size_t)B_ * N_ * N_ * 4);         // 170 MB
    (void)ws_size; (void)in_sizes; (void)n_in; (void)out_size;

    // 1) GroupNorm statistics: one block per (b, group).
    gn_stats_kernel<<<B_ * G_, 256, 0, stream>>>(input, mean, rstd);

    // 2) Weight f32 -> f16.
    cvt_w_kernel<<<CC / 256, 256, 0, stream>>>(Wq, Wk, Wv, w16);

    // 3) Normalize + transpose to pixel-major f16.
    gn_apply_kernel<<<dim3(N_ / 64, C_ / 32, B_), 256, 0, stream>>>(
        input, gn_w, gn_b, mean, rstd, xn16);

    // 4) Q/K/V projections (WMMA, 4x4 blocked).
    qkv_gemm_kernel<<<dim3(N_ / 128, C_ / 128, B_ * 3), 128, 0, stream>>>(
        xn16, w16, bq, bk, bv, q16, k16, v16);

    // 5) Energy E = Q.K^T / 16 (WMMA, 4x4 blocked).
    energy_gemm_kernel<<<dim3(N_ / 128, N_ / 128, B_), 128, 0, stream>>>(
        q16, k16, E32);

    // 6) Row softmax (async-LDS staged, fp32 -> f16 probabilities).
    softmax_kernel<<<B_ * N_, 256, 0, stream>>>(E32, P16);

    // 7) Out = P . V^T (WMMA, 4x4 blocked), fp32 store in [B][C][H][W].
    out_gemm_kernel<<<dim3(N_ / 128, C_ / 128, B_), 128, 0, stream>>>(
        P16, v16, out);
}
